// MatchingModel_34531537060010
// MI455X (gfx1250) — matrix operand order repaired and checked by reading, not compiled, not run on hardware
//
#include <hip/hip_runtime.h>
#include <hip/hip_bf16.h>
#include <math.h>

typedef __attribute__((ext_vector_type(16))) _Float16 v16h;
typedef __attribute__((ext_vector_type(8)))  float    v8f;

#define NPTS   768
#define DMODEL 128
#define NHEAD  4
#define DKH    32
#define FFDIM  512
#define RBFK   16
#define KSIG   5
#define NAUG   769
#define NM     (768*768)
#define TAUINV 2.0f
#define SINK_ITERS 50

__device__ __forceinline__ float gelu_exact(float x) {
    return 0.5f * x * (1.0f + erff(x * 0.7071067811865475f));
}

// |a - b| computed with inline asm, fully opaque to the optimizer.
// Rationale: every source-level absolute value (fabsf, fmax(x,-x), sign-bit
// masking in f32 or f16) is canonicalized by InstCombine back to llvm.fabs.f32,
// and after SLP packs lanes the gfx1250 DAG combiner builds an unselectable
// "v2f32 bitcast(i32 and(i32 bitcast(v2f32 fadd)))" node. The asm block keeps
// the abs out of LLVM IR entirely: v_sub_f32 then a bit-level v_and_b32 sign
// clear. VALU RAW hazards are hardware-interlocked on CDNA5 (s_delay_alu is a
// performance hint only), so the dependent pair needs no scheduling help.
__device__ __forceinline__ float absdiff_f32(float a, float b) {
    float r;
    asm("v_sub_f32 %0, %1, %2\n\t"
        "v_and_b32 %0, 0x7fffffff, %0"
        : "=v"(r)
        : "v"(a), "v"(b));
    return r;
}

// ---------------------------------------------------------------------------
// Generic batched WMMA GEMM:  C = act(alpha * A(f32->f16) * B(f32->f16) + bias + Cadd)
// A: [M,K] row-major (lda).  BLAYOUT 0: B is w[dout,K] row-major -> C = A*B^T.
//                            BLAYOUT 1: B is [K,N] row-major     -> C = A*B.
// One wave per 16x16 C tile; grid = (N/16, M/16, batch).
// Fragment layouts used here (wave32, contiguous k order):
//   A 16x32 f16 : lane m=l16; VGPR v:  K=2v,2v+1 (+16*half)
//   B 32x16 f16 : lane n=l16; VGPR v:  K=2v,2v+1 (+16*half)
//   C 16x16 f32 : lane n=l16; VGPR v:  row = v + 8*half
// ---------------------------------------------------------------------------
template<int BLAYOUT>
__launch_bounds__(32)
__global__ void gemm_wmma_kernel(
    const float* __restrict__ A, int lda, long long sA,
    const float* __restrict__ B, int ldb, long long sB,
    const float* __restrict__ bias,
    const float* __restrict__ Cadd, int ldadd, long long sAdd,
    float* __restrict__ C, int ldc, long long sC,
    int Kdim, float alpha, int act)
{
    const int tn = blockIdx.x, tm = blockIdx.y, bz = blockIdx.z;
    A += sA * bz;  B += sB * bz;  C += sC * bz;
    if (Cadd) Cadd += sAdd * bz;

    const int lane = threadIdx.x & 31;
    const int half = lane >> 4;
    const int l16  = lane & 15;
    const int arow = tm * 16 + l16;
    const int bcol = tn * 16 + l16;

    v8f acc = {};
    for (int kk = 0; kk < Kdim; kk += 32) {
        v16h af, bf;
        const float* Ap = A + (size_t)arow * lda + kk;
        __builtin_prefetch(Ap + 32, 0, 1);
#pragma unroll
        for (int e = 0; e < 16; e += 2) {
            int v  = e >> 1;
            int kb = 2 * v + 16 * half;
            float2 x = *(const float2*)(Ap + kb);
            af[e]     = (_Float16)x.x;
            af[e + 1] = (_Float16)x.y;
        }
        if (BLAYOUT == 0) {
            const float* Bp = B + (size_t)bcol * ldb + kk;
#pragma unroll
            for (int e = 0; e < 16; e += 2) {
                int kb = e + 16 * half;
                float2 x = *(const float2*)(Bp + kb);
                bf[e]     = (_Float16)x.x;
                bf[e + 1] = (_Float16)x.y;
            }
        } else {
            const float* Bp = B + (size_t)kk * ldb + bcol;
#pragma unroll
            for (int e = 0; e < 16; e += 2) {
                int kb = e + 16 * half;
                bf[e]   = (_Float16)Bp[(size_t)kb * ldb];
                bf[e + 1] = (_Float16)Bp[(size_t)(kb + 1) * ldb];
            }
        }
        acc = __builtin_amdgcn_wmma_f32_16x16x32_f16(false, af, false, bf,
                                                     (short)0, acc, false, false);
    }

    const float bc = bias ? bias[bcol] : 0.0f;
#pragma unroll
    for (int v = 0; v < 8; ++v) {
        int row = tm * 16 + v + 8 * half;
        float val = acc[v] * alpha + bc;
        if (Cadd) val += Cadd[(size_t)row * ldadd + bcol];
        if (act == 1) val = gelu_exact(val);
        C[(size_t)row * ldc + bcol] = val;
    }
}

// ---------------------------------------------------------------------------
// Pair MLP: out[i,j] = m2 . gelu(m1 * |HX_i - HY_j| + b1) + b2
// One wave handles 16 rows i for a fixed column j, 8 N-tiles x 4 K-chunks WMMA.
// ---------------------------------------------------------------------------
__launch_bounds__(32)
__global__ void pair_mlp_kernel(const float* __restrict__ HX,
                                const float* __restrict__ HY,
                                const float* __restrict__ m1w,
                                const float* __restrict__ m1b,
                                const float* __restrict__ m2w,
                                const float* __restrict__ m2b,
                                float* __restrict__ out)
{
    const int j    = blockIdx.x;   // 0..767  (HY row)
    const int tm   = blockIdx.y;   // 0..47   (HX row tile)
    const int lane = threadIdx.x & 31;
    const int half = lane >> 4;
    const int l16  = lane & 15;

    __shared__ float hyrow[DMODEL];
    for (int c = lane; c < DMODEL; c += 32) hyrow[c] = HY[(size_t)j * DMODEL + c];
    __syncthreads();

    const int arow = tm * 16 + l16;
    v8f acc[8] = {};
    for (int kk = 0; kk < DMODEL; kk += 32) {
        v16h af;
#pragma unroll
        for (int e = 0; e < 16; e += 2) {
            int v  = e >> 1;
            int kb = 2 * v + 16 * half;
            int k  = kk + kb;
            float2 hx = *(const float2*)(HX + (size_t)arow * DMODEL + k);
            af[e]     = (_Float16)absdiff_f32(hx.x, hyrow[k]);
            af[e + 1] = (_Float16)absdiff_f32(hx.y, hyrow[k + 1]);
        }
#pragma unroll
        for (int t = 0; t < 8; ++t) {
            v16h bf;
            const float* Bp = m1w + (size_t)(t * 16 + l16) * DMODEL + kk;
#pragma unroll
            for (int e = 0; e < 16; e += 2) {
                int kb = e + 16 * half;
                float2 x = *(const float2*)(Bp + kb);
                bf[e]     = (_Float16)x.x;
                bf[e + 1] = (_Float16)x.y;
            }
            acc[t] = __builtin_amdgcn_wmma_f32_16x16x32_f16(false, af, false, bf,
                                                            (short)0, acc[t], false, false);
        }
    }

    float rowsum[8];
#pragma unroll
    for (int v = 0; v < 8; ++v) rowsum[v] = 0.0f;
#pragma unroll
    for (int t = 0; t < 8; ++t) {
        int col = t * 16 + l16;
        float w2 = m2w[col];
        float b1 = m1b[col];
#pragma unroll
        for (int v = 0; v < 8; ++v)
            rowsum[v] += gelu_exact(acc[t][v] + b1) * w2;
    }
    // reduce over the 16 lanes of this half (cols)
#pragma unroll
    for (int m = 8; m >= 1; m >>= 1) {
#pragma unroll
        for (int v = 0; v < 8; ++v)
            rowsum[v] += __shfl_xor(rowsum[v], m, 32);
    }
    if (l16 == 0) {
        float b2 = m2b[0];
#pragma unroll
        for (int v = 0; v < 8; ++v) {
            int row = tm * 16 + v + 8 * half;
            out[(size_t)row * 768 + j] = rowsum[v] + b2;
        }
    }
}

// ---------------------------------------------------------------------------
// Small helper kernels
// ---------------------------------------------------------------------------
__global__ void signature_kernel(const float* __restrict__ P, int n, float* __restrict__ sig)
{
    int i = blockIdx.x * blockDim.x + threadIdx.x;
    if (i >= n) return;
    float px = P[2 * i], py = P[2 * i + 1];
    float best[KSIG];
#pragma unroll
    for (int t = 0; t < KSIG; ++t) best[t] = 1e30f;
    for (int j = 0; j < n; ++j) {
        if (j == i) continue;
        float dx = P[2 * j] - px, dy = P[2 * j + 1] - py;
        float d  = sqrtf(dx * dx + dy * dy + 1e-12f);
        if (d < best[KSIG - 1]) {
            int t = KSIG - 1;
            while (t > 0 && best[t - 1] > d) { best[t] = best[t - 1]; --t; }
            best[t] = d;
        }
    }
#pragma unroll
    for (int t = 0; t < KSIG; ++t) sig[i * KSIG + t] = best[t];
}

__global__ void enc_lin_kernel(const float* __restrict__ sig,
                               const float* __restrict__ w,
                               const float* __restrict__ b,
                               float* __restrict__ out)
{
    int idx = blockIdx.x * blockDim.x + threadIdx.x;
    if (idx >= NPTS * DMODEL) return;
    int r = idx / DMODEL, o = idx % DMODEL;
    float s = b[o];
#pragma unroll
    for (int k = 0; k < KSIG; ++k) s += sig[r * KSIG + k] * w[o * KSIG + k];
    out[idx] = s;
}

__global__ void layernorm_kernel(const float* __restrict__ A,
                                 const float* __restrict__ Radd,
                                 const float* __restrict__ g,
                                 const float* __restrict__ b,
                                 float* __restrict__ out, int applyGelu)
{
    int row = blockIdx.x;
    int t   = threadIdx.x;   // 0..127
    __shared__ float sh[DMODEL];
    float x = A[(size_t)row * DMODEL + t];
    if (Radd) x += Radd[(size_t)row * DMODEL + t];
    sh[t] = x; __syncthreads();
    for (int s = 64; s > 0; s >>= 1) { if (t < s) sh[t] += sh[t + s]; __syncthreads(); }
    float mu = sh[0] * (1.0f / DMODEL);
    __syncthreads();
    float d = x - mu;
    sh[t] = d * d; __syncthreads();
    for (int s = 64; s > 0; s >>= 1) { if (t < s) sh[t] += sh[t + s]; __syncthreads(); }
    float var = sh[0] * (1.0f / DMODEL);
    float y = d / sqrtf(var + 1e-5f) * g[t] + b[t];
    if (applyGelu) y = gelu_exact(y);
    out[(size_t)row * DMODEL + t] = y;
}

__global__ void pdist_kernel(const float* __restrict__ A, const float* __restrict__ B,
                             float* __restrict__ D)
{
    int idx = blockIdx.x * blockDim.x + threadIdx.x;
    if (idx >= NM) return;
    int i = idx / 768, j = idx % 768;
    float dx = A[2 * i] - B[2 * j];
    float dy = A[2 * i + 1] - B[2 * j + 1];
    D[idx] = sqrtf(dx * dx + dy * dy + 1e-12f);
}

__global__ void rbf_bias_kernel(const float* __restrict__ D, int transposeD,
                                const float* __restrict__ centers,
                                const float* __restrict__ log_sigmas,
                                const float* __restrict__ w,
                                float* __restrict__ bias)
{
    int idx = blockIdx.x * blockDim.x + threadIdx.x;
    if (idx >= NM) return;
    int i = idx / 768, j = idx % 768;
    float d = transposeD ? D[(size_t)j * 768 + i] : D[idx];
    float acc[NHEAD] = {0.f, 0.f, 0.f, 0.f};
#pragma unroll
    for (int k = 0; k < RBFK; ++k) {
        float s  = expf(log_sigmas[k]);
        float s2 = s * s + 1e-8f;
        float df = d - centers[k];
        float f  = expf(-df * df / (2.0f * s2));
#pragma unroll
        for (int h = 0; h < NHEAD; ++h) acc[h] += w[h * RBFK + k] * f;
    }
#pragma unroll
    for (int h = 0; h < NHEAD; ++h) bias[(size_t)h * NM + idx] = acc[h];
}

__global__ void softmax_kernel(float* __restrict__ X, int width)
{
    int row = blockIdx.x;
    float* R = X + (size_t)row * width;
    __shared__ float sh[256];
    int t = threadIdx.x;
    float mx = -1e30f;
    for (int c = t; c < width; c += 256) mx = fmaxf(mx, R[c]);
    sh[t] = mx; __syncthreads();
    for (int s = 128; s > 0; s >>= 1) { if (t < s) sh[t] = fmaxf(sh[t], sh[t + s]); __syncthreads(); }
    mx = sh[0]; __syncthreads();
    float sum = 0.0f;
    for (int c = t; c < width; c += 256) { float e = expf(R[c] - mx); R[c] = e; sum += e; }
    sh[t] = sum; __syncthreads();
    for (int s = 128; s > 0; s >>= 1) { if (t < s) sh[t] += sh[t + s]; __syncthreads(); }
    float inv = 1.0f / sh[0];
    for (int c = t; c < width; c += 256) R[c] *= inv;
}

__global__ void dust_kernel(const float* __restrict__ H,
                            const float* __restrict__ w1, const float* __restrict__ b1,
                            const float* __restrict__ w2, const float* __restrict__ b2,
                            float* __restrict__ out)
{
    int row = blockIdx.x;
    int t   = threadIdx.x;   // 0..63
    __shared__ float sh[64];
    float s = b1[t];
    for (int k = 0; k < DMODEL; ++k) s += H[(size_t)row * DMODEL + k] * w1[t * DMODEL + k];
    sh[t] = gelu_exact(s) * w2[t];
    __syncthreads();
    for (int st = 32; st > 0; st >>= 1) { if (t < st) sh[t] += sh[t + st]; __syncthreads(); }
    if (t == 0) out[row] = sh[0] + b2[0];
}

__global__ void build_aug_kernel(const float* __restrict__ sim, const float* __restrict__ mlp,
                                 const float* __restrict__ dx, const float* __restrict__ dy,
                                 const float* __restrict__ pa, const float* __restrict__ pb,
                                 const float* __restrict__ ps, float* __restrict__ Z)
{
    int idx = blockIdx.x * blockDim.x + threadIdx.x;
    if (idx >= NAUG * NAUG) return;
    int i = idx / NAUG, j = idx % NAUG;
    float a = pa[0], b = pb[0], sc = ps[0];
    float v;
    if (i < 768 && j < 768)      v = (a * sim[(size_t)i * 768 + j] + b * mlp[(size_t)i * 768 + j]) * sc;
    else if (i < 768)            v = dx[i];
    else if (j < 768)            v = dy[j];
    else                         v = 0.0f;
    Z[idx] = v * TAUINV;
}

__global__ void lse_rows_kernel(float* __restrict__ Z)
{
    int r = blockIdx.x;
    float* R = Z + (size_t)r * NAUG;
    __shared__ float sh[256];
    int t = threadIdx.x;
    float mx = -1e30f;
    for (int c = t; c < NAUG; c += 256) mx = fmaxf(mx, R[c]);
    sh[t] = mx; __syncthreads();
    for (int s = 128; s > 0; s >>= 1) { if (t < s) sh[t] = fmaxf(sh[t], sh[t + s]); __syncthreads(); }
    mx = sh[0]; __syncthreads();
    float sum = 0.0f;
    for (int c = t; c < NAUG; c += 256) sum += expf(R[c] - mx);
    sh[t] = sum; __syncthreads();
    for (int s = 128; s > 0; s >>= 1) { if (t < s) sh[t] += sh[t + s]; __syncthreads(); }
    float lse = mx + logf(sh[0]);
    for (int c = t; c < NAUG; c += 256) R[c] -= lse;
}

__global__ void lse_cols_kernel(float* __restrict__ Z)
{
    int c = blockIdx.x * blockDim.x + threadIdx.x;
    if (c >= NAUG) return;
    float mx = -1e30f;
    for (int r = 0; r < NAUG; ++r) mx = fmaxf(mx, Z[(size_t)r * NAUG + c]);
    float s = 0.0f;
    for (int r = 0; r < NAUG; ++r) s += expf(Z[(size_t)r * NAUG + c] - mx);
    float lse = mx + logf(s);
    for (int r = 0; r < NAUG; ++r) Z[(size_t)r * NAUG + c] -= lse;
}

__global__ void exp_kernel(const float* __restrict__ Z, float* __restrict__ out, int n)
{
    int i = blockIdx.x * blockDim.x + threadIdx.x;
    if (i < n) out[i] = expf(Z[i]);
}

// ---------------------------------------------------------------------------
// Host-side orchestration
// ---------------------------------------------------------------------------
struct LinP  { const float* w; const float* b; };
struct LNP   { const float* g; const float* b; };
struct AttnP { LinP q, k, v, o; };
struct BlockP{ AttnP attn; LNP ln1; LinP f1, f2; LNP ln2; };
struct RbfP  { const float* centers; const float* log_sigmas; const float* w; };

extern "C" void kernel_launch(void* const* d_in, const int* in_sizes, int n_in,
                              void* d_out, int out_size, void* d_ws, size_t ws_size,
                              hipStream_t stream)
{
    (void)in_sizes; (void)n_in; (void)out_size; (void)ws_size;

    // ---- parse inputs in setup_inputs() flattening order ----
    int cur = 0;
    auto next = [&]() -> const float* { return (const float*)d_in[cur++]; };
    const float* Xc = next();
    const float* Yc = next();
    auto readLin  = [&]() { LinP l; l.w = next(); l.b = next(); return l; };
    auto readLN   = [&]() { LNP  l; l.g = next(); l.b = next(); return l; };
    auto readRbf  = [&]() { RbfP r; r.centers = next(); r.log_sigmas = next(); r.w = next(); return r; };
    auto readAttn = [&]() { AttnP a; a.q = readLin(); a.k = readLin(); a.v = readLin(); a.o = readLin(); return a; };
    auto readBlock= [&]() { BlockP b; b.attn = readAttn(); b.ln1 = readLN();
                            b.f1 = readLin(); b.f2 = readLin(); b.ln2 = readLN(); return b; };

    LinP enc_lin = readLin();
    LNP  enc_ln  = readLN();
    RbfP rxx = readRbf(), ryy = readRbf(), rxy = readRbf(), ryx = readRbf();
    BlockP self_x[4], self_y[4], cxy[4], cyx[4];
    for (int l = 0; l < 4; ++l) self_x[l] = readBlock();
    for (int l = 0; l < 4; ++l) self_y[l] = readBlock();
    for (int l = 0; l < 4; ++l) { cxy[l] = readBlock(); cyx[l] = readBlock(); }
    const float* pscale = next();
    const float* palpha = next();
    const float* pbeta  = next();
    LinP m1 = readLin(), m2 = readLin();
    LinP mx1 = readLin(), mx2 = readLin(), my1 = readLin(), my2 = readLin();

    // ---- workspace carve (256-float aligned slices) ----
    float* ws = (float*)d_ws;
    size_t off = 0;
    auto alloc = [&](size_t n) { float* p = ws + off; off += (n + 255) & ~(size_t)255; return p; };
    float* sigX = alloc(NPTS * KSIG);
    float* sigY = alloc(NPTS * KSIG);
    float* HX   = alloc(NPTS * DMODEL);
    float* HY   = alloc(NPTS * DMODEL);
    float* Dxx  = alloc(NM);
    float* Dyy  = alloc(NM);
    float* Dxy  = alloc(NM);
    float* bXX  = alloc((size_t)NHEAD * NM);
    float* bYY  = alloc((size_t)NHEAD * NM);
    float* bXY  = alloc((size_t)NHEAD * NM);
    float* bYX  = alloc((size_t)NHEAD * NM);
    float* ATT  = alloc((size_t)NHEAD * NM);
    float* Qb   = alloc(NPTS * DMODEL);
    float* Kb   = alloc(NPTS * DMODEL);
    float* Vb   = alloc(NPTS * DMODEL);
    float* AO   = alloc(NPTS * DMODEL);
    float* T1   = alloc(NPTS * DMODEL);
    float* H1   = alloc(NPTS * DMODEL);
    float* FFH  = alloc(NPTS * FFDIM);
    float* SIM  = alloc(NM);
    float* MLP  = alloc(NM);
    float* DX   = alloc(NPTS);
    float* DY   = alloc(NPTS);
    float* Z    = alloc((size_t)NAUG * NAUG);

    const float inv_sqrt_dk = 0.17677669529663687f;   // 1/sqrt(32)
    const float inv_sqrt_d  = 0.08838834764831845f;   // 1/sqrt(128)

    auto gemmNK = [&](const float* A, int lda, long long sA,
                      const float* B, int ldb, long long sB,
                      const float* bias, const float* Cadd, int ldadd, long long sAdd,
                      float* C, int ldc, long long sC,
                      int Mr, int Nc, int Kd, float alpha, int act, int batch) {
        dim3 g(Nc / 16, Mr / 16, batch);
        gemm_wmma_kernel<0><<<g, 32, 0, stream>>>(A, lda, sA, B, ldb, sB, bias,
                                                  Cadd, ldadd, sAdd, C, ldc, sC, Kd, alpha, act);
    };
    auto gemmKN = [&](const float* A, int lda, long long sA,
                      const float* B, int ldb, long long sB,
                      float* C, int ldc, long long sC,
                      int Mr, int Nc, int Kd, float alpha, int batch) {
        dim3 g(Nc / 16, Mr / 16, batch);
        gemm_wmma_kernel<1><<<g, 32, 0, stream>>>(A, lda, sA, B, ldb, sB, nullptr,
                                                  nullptr, 0, 0, C, ldc, sC, Kd, alpha, 0);
    };

    auto mha_block = [&](const BlockP& blk, const float* Ain, const float* Bin,
                         const float* bias4, float* Hout) {
        // projections
        gemmNK(Ain, DMODEL, 0, blk.attn.q.w, DMODEL, 0, blk.attn.q.b, nullptr, 0, 0,
               Qb, DMODEL, 0, NPTS, DMODEL, DMODEL, 1.0f, 0, 1);
        gemmNK(Bin, DMODEL, 0, blk.attn.k.w, DMODEL, 0, blk.attn.k.b, nullptr, 0, 0,
               Kb, DMODEL, 0, NPTS, DMODEL, DMODEL, 1.0f, 0, 1);
        gemmNK(Bin, DMODEL, 0, blk.attn.v.w, DMODEL, 0, blk.attn.v.b, nullptr, 0, 0,
               Vb, DMODEL, 0, NPTS, DMODEL, DMODEL, 1.0f, 0, 1);
        // logits[h] = Qh * Kh^T / sqrt(dk) + bias[h]
        gemmNK(Qb, DMODEL, DKH, Kb, DMODEL, DKH, nullptr, bias4, 768, NM,
               ATT, 768, NM, NPTS, NPTS, DKH, inv_sqrt_dk, 0, NHEAD);
        softmax_kernel<<<NHEAD * NPTS, 256, 0, stream>>>(ATT, 768);
        // out[h] = attn[h] * Vh
        gemmKN(ATT, 768, NM, Vb, DMODEL, DKH, AO, DMODEL, DKH,
               NPTS, DKH, NPTS, 1.0f, NHEAD);
        // output projection
        gemmNK(AO, DMODEL, 0, blk.attn.o.w, DMODEL, 0, blk.attn.o.b, nullptr, 0, 0,
               T1, DMODEL, 0, NPTS, DMODEL, DMODEL, 1.0f, 0, 1);
        // H1 = LN1(Ain + T1)
        layernorm_kernel<<<NPTS, DMODEL, 0, stream>>>(T1, Ain, blk.ln1.g, blk.ln1.b, H1, 0);
        // FFN
        gemmNK(H1, DMODEL, 0, blk.f1.w, DMODEL, 0, blk.f1.b, nullptr, 0, 0,
               FFH, FFDIM, 0, NPTS, FFDIM, DMODEL, 1.0f, 1, 1);
        gemmNK(FFH, FFDIM, 0, blk.f2.w, FFDIM, 0, blk.f2.b, nullptr, 0, 0,
               T1, DMODEL, 0, NPTS, DMODEL, FFDIM, 1.0f, 0, 1);
        // Hout = LN2(H1 + T1)
        layernorm_kernel<<<NPTS, DMODEL, 0, stream>>>(T1, H1, blk.ln2.g, blk.ln2.b, Hout, 0);
    };

    // ---- encoder ----
    signature_kernel<<<3, 256, 0, stream>>>(Xc, NPTS, sigX);
    signature_kernel<<<3, 256, 0, stream>>>(Yc, NPTS, sigY);
    enc_lin_kernel<<<(NPTS * DMODEL) / 256, 256, 0, stream>>>(sigX, enc_lin.w, enc_lin.b, T1);
    layernorm_kernel<<<NPTS, DMODEL, 0, stream>>>(T1, nullptr, enc_ln.g, enc_ln.b, HX, 1);
    enc_lin_kernel<<<(NPTS * DMODEL) / 256, 256, 0, stream>>>(sigY, enc_lin.w, enc_lin.b, T1);
    layernorm_kernel<<<NPTS, DMODEL, 0, stream>>>(T1, nullptr, enc_ln.g, enc_ln.b, HY, 1);

    // ---- distances + RBF biases (precomputed once, reused by all 4 layers; live in L2) ----
    pdist_kernel<<<NM / 256, 256, 0, stream>>>(Xc, Xc, Dxx);
    pdist_kernel<<<NM / 256, 256, 0, stream>>>(Yc, Yc, Dyy);
    pdist_kernel<<<NM / 256, 256, 0, stream>>>(Xc, Yc, Dxy);
    rbf_bias_kernel<<<NM / 256, 256, 0, stream>>>(Dxx, 0, rxx.centers, rxx.log_sigmas, rxx.w, bXX);
    rbf_bias_kernel<<<NM / 256, 256, 0, stream>>>(Dyy, 0, ryy.centers, ryy.log_sigmas, ryy.w, bYY);
    rbf_bias_kernel<<<NM / 256, 256, 0, stream>>>(Dxy, 0, rxy.centers, rxy.log_sigmas, rxy.w, bXY);
    rbf_bias_kernel<<<NM / 256, 256, 0, stream>>>(Dxy, 1, ryx.centers, ryx.log_sigmas, ryx.w, bYX);

    // ---- transformer stack ----
    for (int l = 0; l < 4; ++l) {
        mha_block(self_x[l], HX, HX, bXX, HX);
        mha_block(self_y[l], HY, HY, bYY, HY);
        mha_block(cxy[l],    HX, HY, bXY, HX);
        mha_block(cyx[l],    HY, HX, bYX, HY);
    }

    // ---- pairwise scores ----
    gemmNK(HX, DMODEL, 0, HY, DMODEL, 0, nullptr, nullptr, 0, 0,
           SIM, 768, 0, NPTS, NPTS, DMODEL, inv_sqrt_d, 0, 1);
    pair_mlp_kernel<<<dim3(768, 48, 1), 32, 0, stream>>>(HX, HY, m1.w, m1.b, m2.w, m2.b, MLP);
    dust_kernel<<<NPTS, 64, 0, stream>>>(HX, mx1.w, mx1.b, mx2.w, mx2.b, DX);
    dust_kernel<<<NPTS, 64, 0, stream>>>(HY, my1.w, my1.b, my2.w, my2.b, DY);

    // ---- Sinkhorn ----
    build_aug_kernel<<<(NAUG * NAUG + 255) / 256, 256, 0, stream>>>(SIM, MLP, DX, DY,
                                                                    palpha, pbeta, pscale, Z);
    for (int it = 0; it < SINK_ITERS; ++it) {
        lse_rows_kernel<<<NAUG, 256, 0, stream>>>(Z);
        lse_cols_kernel<<<(NAUG + 255) / 256, 256, 0, stream>>>(Z);
    }
    exp_kernel<<<(NAUG * NAUG + 255) / 256, 256, 0, stream>>>(Z, (float*)d_out, NAUG * NAUG);
}
